// MultiHeadSelfAttention_52673478918821
// MI455X (gfx1250) — matrix operand order repaired and checked by reading, not compiled, not run on hardware
//
#include <hip/hip_runtime.h>
#include <hip/hip_bf16.h>

typedef __bf16 bf16;
typedef __attribute__((ext_vector_type(16))) __bf16 v16bf;
typedef __attribute__((ext_vector_type(8)))  __bf16 bf16x8;
typedef __attribute__((ext_vector_type(8)))  float  v8f;

static __device__ __forceinline__ v8f zero8() {
  v8f z;
#pragma unroll
  for (int i = 0; i < 8; ++i) z[i] = 0.0f;
  return z;
}

static __device__ __forceinline__ v8f wmma_bf16(v16bf a, v16bf b, v8f c) {
  return __builtin_amdgcn_wmma_f32_16x16x32_bf16(false, a, false, b, (short)0, c,
                                                 false, false);
}

// A fragment (16x32 bf16) from a row-major tile, leading dim `ld` elements.
// ISA: lanes 0-15 row=lane, K = 0..7 & 16..23; lanes 16-31 K = 8..15 & 24..31.
static __device__ __forceinline__ v16bf load_a_frag(const bf16* tile, int ld, int lane) {
  const bf16* p = tile + (lane & 15) * ld + ((lane >> 4) << 3);
  bf16x8 lo = *(const bf16x8*)(p);
  bf16x8 hi = *(const bf16x8*)(p + 16);
  v16bf a;
#pragma unroll
  for (int i = 0; i < 8; ++i) { a[i] = lo[i]; a[8 + i] = hi[i]; }
  return a;
}

// B fragment (32x16 bf16) from an n-major (transposed) tile: row = n, contiguous k.
// ISA: lane = n (mod 16); lanes 0-15 K = 0..7 & 16..23; lanes 16-31 K = 8..15 & 24..31.
static __device__ __forceinline__ v16bf load_b_frag(const bf16* tile, int ld, int lane) {
  const bf16* p = tile + (lane & 15) * ld + ((lane >> 4) << 3);
  bf16x8 lo = *(const bf16x8*)(p);
  bf16x8 hi = *(const bf16x8*)(p + 16);
  v16bf b;
#pragma unroll
  for (int i = 0; i < 8; ++i) { b[i] = lo[i]; b[8 + i] = hi[i]; }
  return b;
}

// -------------------- Kernel 1: QKV projection --------------------
// X[8192,1024] fp32 x W[1024,3072] fp32 + bias -> Q/K/V bf16 [B,H,T,64]
// (K is pre-scaled by 1/sqrt(64) here so attention needs no score scaling.)
__global__ __launch_bounds__(128)
void qkv_gemm_kernel(const float* __restrict__ X, const float* __restrict__ W,
                     const float* __restrict__ bias, bf16* __restrict__ Qb,
                     bf16* __restrict__ Kb, bf16* __restrict__ Vb) {
  __shared__ bf16 As[128][40];
  __shared__ bf16 Bs[64][40];
  const int tid = threadIdx.x;
  const int lane = tid & 31;
  const int w = tid >> 5;
  const int wm = w >> 1, wn = w & 1;
  const int Nb = blockIdx.x * 64;
  const int Mb = blockIdx.y * 128;

  v8f acc[4][2];
#pragma unroll
  for (int i = 0; i < 4; ++i)
#pragma unroll
    for (int j = 0; j < 2; ++j) acc[i][j] = zero8();

  const float* xrow = X + (size_t)(Mb + tid) * 1024;
  const int bn = tid & 63;          // B-tile: this thread owns column n=Nb+bn
  const int bk = (tid >> 6) * 16;   // and k rows [bk, bk+16)

  float4 aReg[8];
  float  bReg[16];

  // ---- prologue: load tile k0=0 into registers ----
#pragma unroll
  for (int j = 0; j < 8; ++j) aReg[j] = ((const float4*)(xrow))[j];
#pragma unroll
  for (int j = 0; j < 16; ++j)
    bReg[j] = W[(size_t)(bk + j) * 3072 + Nb + bn];

  for (int kt = 0; kt < 32; ++kt) {
    __syncthreads();
    {  // packed bf16 stores: 4x ds_store_b128 for A, 2x for B
#pragma unroll
      for (int j = 0; j < 4; ++j) {
        bf16x8 p;
        p[0] = (bf16)aReg[2 * j].x;     p[1] = (bf16)aReg[2 * j].y;
        p[2] = (bf16)aReg[2 * j].z;     p[3] = (bf16)aReg[2 * j].w;
        p[4] = (bf16)aReg[2 * j + 1].x; p[5] = (bf16)aReg[2 * j + 1].y;
        p[6] = (bf16)aReg[2 * j + 1].z; p[7] = (bf16)aReg[2 * j + 1].w;
        *(bf16x8*)(&As[tid][j * 8]) = p;
      }
#pragma unroll
      for (int j = 0; j < 2; ++j) {
        bf16x8 p;
#pragma unroll
        for (int e = 0; e < 8; ++e) p[e] = (bf16)bReg[j * 8 + e];
        *(bf16x8*)(&Bs[bn][bk + j * 8]) = p;
      }
    }
    __syncthreads();

    if (kt < 31) {  // pipeline: fetch next tile while WMMAs run
      int k0 = (kt + 1) * 32;
#pragma unroll
      for (int j = 0; j < 8; ++j) aReg[j] = ((const float4*)(xrow + k0))[j];
#pragma unroll
      for (int j = 0; j < 16; ++j)
        bReg[j] = W[(size_t)(k0 + bk + j) * 3072 + Nb + bn];
    }

    v16bf afr[4];
#pragma unroll
    for (int mi = 0; mi < 4; ++mi)
      afr[mi] = load_a_frag(&As[wm * 64 + mi * 16][0], 40, lane);
    v16bf bfr[2];
#pragma unroll
    for (int ni = 0; ni < 2; ++ni)
      bfr[ni] = load_b_frag(&Bs[wn * 32 + ni * 16][0], 40, lane);
#pragma unroll
    for (int mi = 0; mi < 4; ++mi)
#pragma unroll
      for (int ni = 0; ni < 2; ++ni)
        acc[mi][ni] = wmma_bf16(afr[mi], bfr[ni], acc[mi][ni]);
  }

  const int lh = lane >> 4;
  const int nl = lane & 15;
#pragma unroll
  for (int mi = 0; mi < 4; ++mi) {
#pragma unroll
    for (int ni = 0; ni < 2; ++ni) {
      int n = Nb + wn * 32 + ni * 16 + nl;
      float bv = bias[n];
      int which = n >> 10;
      int h = (n & 1023) >> 6;
      int d = n & 63;
      bf16* dst = (which == 0) ? Qb : (which == 1) ? Kb : Vb;
      float sc = (which == 1) ? 0.125f : 1.0f;  // fold 1/sqrt(hd) into K
#pragma unroll
      for (int i = 0; i < 8; ++i) {
        int m = Mb + wm * 64 + mi * 16 + i + lh * 8;
        int bb = m >> 11;
        int t = m & 2047;
        float val = (acc[mi][ni][i] + bv) * sc;
        dst[((size_t)(bb * 16 + h) * 2048 + t) * 64 + d] = (bf16)val;
      }
    }
  }
}

// -------------------- Kernel 2: flash attention --------------------
__global__ __launch_bounds__(128)
void attn_kernel(const bf16* __restrict__ Qb, const bf16* __restrict__ Kb,
                 const bf16* __restrict__ Vb, const int* __restrict__ mask,
                 bf16* __restrict__ Ab) {
  __shared__ bf16 Vt[64][72];     // V transposed: [d][k]
  __shared__ bf16 Ps[4][16][72];  // per-wave P staging (D-layout -> A-layout)

  const int tid = threadIdx.x;
  const int lane = tid & 31;
  const int w = tid >> 5;
  const int q0 = blockIdx.x * 64;
  const int h = blockIdx.y;
  const int b = blockIdx.z;
  const size_t bh = (size_t)(b * 16 + h);
  const bf16* Qbase = Qb + bh * 2048 * 64;
  const bf16* Kbase = Kb + bh * 2048 * 64;
  const bf16* Vbase = Vb + bh * 2048 * 64;

  const int lh = lane >> 4;
  const int nl = lane & 15;

  // Q A-fragments straight from global (row-major [T,64] matches A layout)
  v16bf qa0, qa1;
  {
    const bf16* qp = Qbase + (size_t)(q0 + w * 16 + nl) * 64 + (lh << 3);
    bf16x8 q00 = *(const bf16x8*)(qp);
    bf16x8 q01 = *(const bf16x8*)(qp + 16);
    bf16x8 q10 = *(const bf16x8*)(qp + 32);
    bf16x8 q11 = *(const bf16x8*)(qp + 48);
#pragma unroll
    for (int i = 0; i < 8; ++i) {
      qa0[i] = q00[i]; qa0[8 + i] = q01[i];
      qa1[i] = q10[i]; qa1[8 + i] = q11[i];
    }
  }

  v8f o_acc[4];
#pragma unroll
  for (int i = 0; i < 4; ++i) o_acc[i] = zero8();
  float rmax[8], rsum[8];
#pragma unroll
  for (int i = 0; i < 8; ++i) { rmax[i] = -3.0e38f; rsum[i] = 0.0f; }

  // pipeline regs for V tile (this thread: row vk, 32 d starting at vd0)
  const int vk = tid >> 1;
  const int vd0 = (tid & 1) * 32;
  bf16x8 vreg[4];
  {
    const bf16x8* src = (const bf16x8*)(Vbase + (size_t)vk * 64 + vd0);
#pragma unroll
    for (int j = 0; j < 4; ++j) vreg[j] = src[j];
  }

  for (int kv0 = 0; kv0 < 2048; kv0 += 64) {
    __syncthreads();
    {  // scatter V regs transposed into LDS
#pragma unroll
      for (int j = 0; j < 4; ++j)
#pragma unroll
        for (int e = 0; e < 8; ++e) Vt[vd0 + j * 8 + e][vk] = vreg[j][e];
    }
    __syncthreads();

    if (kv0 + 64 < 2048) {  // pipeline next V tile; prefetch next K block
      const bf16x8* src =
          (const bf16x8*)(Vbase + (size_t)(kv0 + 64 + vk) * 64 + vd0);
#pragma unroll
      for (int j = 0; j < 4; ++j) vreg[j] = src[j];
      __builtin_prefetch(Kbase + (size_t)(kv0 + 64) * 64 + (tid << 5), 0, 0);
    }

    // S = Q . K^T : B-frag columns = K rows (contiguous d in memory)
    v8f s[4];
#pragma unroll
    for (int nt = 0; nt < 4; ++nt) {
      int kvcol = kv0 + nt * 16 + nl;
      const bf16* kp = Kbase + (size_t)kvcol * 64 + lh * 8;
      bf16x8 k0v = *(const bf16x8*)(kp);
      bf16x8 k1v = *(const bf16x8*)(kp + 16);
      bf16x8 k2v = *(const bf16x8*)(kp + 32);
      bf16x8 k3v = *(const bf16x8*)(kp + 48);
      v16bf b0, b1;
#pragma unroll
      for (int i = 0; i < 8; ++i) {
        b0[i] = k0v[i]; b0[8 + i] = k1v[i];
        b1[i] = k2v[i]; b1[8 + i] = k3v[i];
      }
      s[nt] = wmma_bf16(qa0, b0, zero8());
      s[nt] = wmma_bf16(qa1, b1, s[nt]);
      if (mask[b * 2048 + kvcol] == 0) {
#pragma unroll
        for (int i = 0; i < 8; ++i) s[nt][i] = -1.0e9f;
      }
    }

    // online softmax; row m of a D tile lives in one VGPR across a 16-lane half
    float alpha[8];
#pragma unroll
    for (int i = 0; i < 8; ++i) {
      float cm = fmaxf(fmaxf(s[0][i], s[1][i]), fmaxf(s[2][i], s[3][i]));
#pragma unroll
      for (int off = 1; off <= 8; off <<= 1)
        cm = fmaxf(cm, __shfl_xor(cm, off, 16));
      float mnew = fmaxf(rmax[i], cm);
      alpha[i] = __expf(rmax[i] - mnew);
      rmax[i] = mnew;
#pragma unroll
      for (int nt = 0; nt < 4; ++nt) s[nt][i] = __expf(s[nt][i] - mnew);
      float ps = s[0][i] + s[1][i] + s[2][i] + s[3][i];
#pragma unroll
      for (int off = 1; off <= 8; off <<= 1)
        ps += __shfl_xor(ps, off, 16);
      rsum[i] = rsum[i] * alpha[i] + ps;
    }
#pragma unroll
    for (int nt = 0; nt < 4; ++nt)
#pragma unroll
      for (int i = 0; i < 8; ++i) o_acc[nt][i] *= alpha[i];

    // P: D-layout regs -> per-wave LDS -> A-layout frags
#pragma unroll
    for (int nt = 0; nt < 4; ++nt)
#pragma unroll
      for (int i = 0; i < 8; ++i)
        Ps[w][i + lh * 8][nt * 16 + nl] = (bf16)s[nt][i];

    v16bf pa0 = load_a_frag(&Ps[w][0][0], 72, lane);
    v16bf pa1 = load_a_frag(&Ps[w][0][32], 72, lane);
#pragma unroll
    for (int nt2 = 0; nt2 < 4; ++nt2) {
      v16bf vb0 = load_b_frag(&Vt[nt2 * 16][0], 72, lane);
      v16bf vb1 = load_b_frag(&Vt[nt2 * 16][32], 72, lane);
      o_acc[nt2] = wmma_bf16(pa0, vb0, o_acc[nt2]);
      o_acc[nt2] = wmma_bf16(pa1, vb1, o_acc[nt2]);
    }
  }

  // normalize and write attention output as [B, T, H, 64] = [8192,1024] bf16
#pragma unroll
  for (int nt2 = 0; nt2 < 4; ++nt2) {
#pragma unroll
    for (int i = 0; i < 8; ++i) {
      int ml = i + lh * 8;
      int t = q0 + w * 16 + ml;
      int d = nt2 * 16 + nl;
      float val = o_acc[nt2][i] / rsum[i];
      Ab[((size_t)(b * 2048 + t) * 16 + h) * 64 + d] = (bf16)val;
    }
  }
}

// -------------------- Kernel 3: output projection --------------------
// A-tile (pure bf16 copy) uses CDNA5 async global->LDS DMA; B-tile converted
// through registers.
__global__ __launch_bounds__(128)
void out_gemm_kernel(const bf16* __restrict__ A, const float* __restrict__ W,
                     const float* __restrict__ bias, float* __restrict__ out) {
  __shared__ bf16 As[128][40];
  __shared__ bf16 Bs[64][40];
  const int tid = threadIdx.x;
  const int lane = tid & 31;
  const int w = tid >> 5;
  const int wm = w >> 1, wn = w & 1;
  const int Nb = blockIdx.x * 64;
  const int Mb = blockIdx.y * 128;

  v8f acc[4][2];
#pragma unroll
  for (int i = 0; i < 4; ++i)
#pragma unroll
    for (int j = 0; j < 2; ++j) acc[i][j] = zero8();

  const bf16* arow = A + (size_t)(Mb + tid) * 1024;
  const unsigned ldsA = (unsigned)(uintptr_t)(&As[tid][0]);
  const int bn = tid & 63;
  const int bk = (tid >> 6) * 16;

  for (int k0 = 0; k0 < 1024; k0 += 32) {
    __syncthreads();  // previous tile fully consumed
    {  // A tile: 4x async b128 copies global -> LDS (ASYNCcnt tracked)
      unsigned long long ga = (unsigned long long)(uintptr_t)(arow + k0);
#pragma unroll
      for (int j = 0; j < 4; ++j) {
        asm volatile("global_load_async_to_lds_b128 %0, %1, off"
                     :
                     : "v"(ldsA + j * 16), "v"(ga + j * 16)
                     : "memory");
      }
    }
    {  // B tile: per-thread one n-column, 16 k rows; packed b128 LDS stores
      float bReg[16];
#pragma unroll
      for (int j = 0; j < 16; ++j)
        bReg[j] = W[(size_t)(k0 + bk + j) * 1024 + Nb + bn];
#pragma unroll
      for (int j = 0; j < 2; ++j) {
        bf16x8 p;
#pragma unroll
        for (int e = 0; e < 8; ++e) p[e] = (bf16)bReg[j * 8 + e];
        *(bf16x8*)(&Bs[bn][bk + j * 8]) = p;
      }
    }
    asm volatile("s_wait_asynccnt 0" ::: "memory");
    __syncthreads();

    v16bf afr[4];
#pragma unroll
    for (int mi = 0; mi < 4; ++mi)
      afr[mi] = load_a_frag(&As[wm * 64 + mi * 16][0], 40, lane);
    v16bf bfr[2];
#pragma unroll
    for (int ni = 0; ni < 2; ++ni)
      bfr[ni] = load_b_frag(&Bs[wn * 32 + ni * 16][0], 40, lane);
#pragma unroll
    for (int mi = 0; mi < 4; ++mi)
#pragma unroll
      for (int ni = 0; ni < 2; ++ni)
        acc[mi][ni] = wmma_bf16(afr[mi], bfr[ni], acc[mi][ni]);
  }

  const int lh = lane >> 4;
  const int nl = lane & 15;
#pragma unroll
  for (int mi = 0; mi < 4; ++mi) {
#pragma unroll
    for (int ni = 0; ni < 2; ++ni) {
      int n = Nb + wn * 32 + ni * 16 + nl;
      float bv = bias[n];
#pragma unroll
      for (int i = 0; i < 8; ++i) {
        int m = Mb + wm * 64 + mi * 16 + i + lh * 8;
        out[(size_t)m * 1024 + n] = acc[mi][ni][i] + bv;
      }
    }
  }
}

extern "C" void kernel_launch(void* const* d_in, const int* in_sizes, int n_in,
                              void* d_out, int out_size, void* d_ws, size_t ws_size,
                              hipStream_t stream) {
  const float* x     = (const float*)d_in[0];
  const int*   mask  = (const int*)d_in[1];
  const float* W_qkv = (const float*)d_in[2];
  const float* b_qkv = (const float*)d_in[3];
  const float* W_out = (const float*)d_in[4];
  const float* b_out = (const float*)d_in[5];
  float* out = (float*)d_out;

  const size_t elems = (size_t)4 * 16 * 2048 * 64;  // 8388608 per tensor
  bf16* Qb = (bf16*)d_ws;
  bf16* Kb = Qb + elems;
  bf16* Vb = Kb + elems;
  bf16* Ab = Vb + elems;  // total 64 MB bf16 workspace

  dim3 g1(3072 / 64, 8192 / 128);
  qkv_gemm_kernel<<<g1, 128, 0, stream>>>(x, W_qkv, b_qkv, Qb, Kb, Vb);

  dim3 g2(2048 / 64, 16, 4);
  attn_kernel<<<g2, 128, 0, stream>>>(Qb, Kb, Vb, mask, Ab);

  dim3 g3(1024 / 64, 8192 / 128);
  out_gemm_kernel<<<g3, 128, 0, stream>>>(Ab, W_out, b_out, out);
}
